// BConv2d_35510789603655
// MI455X (gfx1250) — matrix operand order repaired and checked by reading, not compiled, not run on hardware
//
#include <hip/hip_runtime.h>

// ---------- types for WMMA ----------
typedef __attribute__((ext_vector_type(8)))  unsigned short ushort8;
typedef __attribute__((ext_vector_type(16))) unsigned short ushort16;
typedef __attribute__((ext_vector_type(16))) __bf16         v16bf;
typedef __attribute__((ext_vector_type(8)))  float          v8f;

#define CIN    32
#define COUT   32
#define HSZ    256
#define WSZ    256
#define NBATCH 32
#define WT     128          // pixels (w) per block
#define HT     2            // output rows per block
#define SROWS  (HT + 2)     // staged input rows (with vertical halo)
#define SWID   (WT + 2)     // staged width (with horizontal halo) = 130
#define XPITCH 40           // padded channel pitch (shorts): 80B, multiple of 16B

// f32 -> bf16, round to nearest even
__device__ __forceinline__ unsigned short f2bf(float f) {
  union { float f; unsigned u; } v; v.f = f;
  unsigned u = v.u;
  u += 0x7FFFu + ((u >> 16) & 1u);
  return (unsigned short)(u >> 16);
}

// build a 16-element bf16 fragment from two 16B LDS chunks (second at +off shorts)
__device__ __forceinline__ v16bf frag16(const unsigned short* p, int off) {
  ushort8 lo = *(const ushort8*)p;
  ushort8 hi = *(const ushort8*)(p + off);
  ushort16 u = __builtin_shufflevector(lo, hi, 0,1,2,3,4,5,6,7,8,9,10,11,12,13,14,15);
  return __builtin_bit_cast(v16bf, u);
}

__global__ __launch_bounds__(256)
void bconv3x3_bin_wmma(const float* __restrict__ x,
                       const float* __restrict__ wgt,
                       float* __restrict__ out) {
  // [tap(9)][co(32)][ci(32)] binarized bf16 weights
  __shared__ unsigned short sW[9 * COUT * CIN];
  // [row(4)][wloc(130)][ci(32) in pitch 40] bf16 input tile with zero halo
  __shared__ unsigned short sX[SROWS * SWID * XPITCH];

  const int tid = threadIdx.x;
  const int w0  = blockIdx.x * WT;
  const int h0  = blockIdx.y * HT;
  const int n   = blockIdx.z;

  // ---- stage binarized weights: weight[co][ci][kh][kw] -> sW[kk][co][ci] ----
  for (int i = tid; i < 9 * COUT * CIN; i += 256) {
    int co = i / (CIN * 9);
    int r  = i - co * (CIN * 9);
    int ci = r / 9;
    int kk = r - ci * 9;
    float v = wgt[i];
    sW[(kk * COUT + co) * CIN + ci] =
        (v > 0.f) ? (unsigned short)0x3F80 : (unsigned short)0xBF80; // +1 / -1 bf16
  }

  // ---- stage input rows as bf16 with zero-padded halo ----
  for (int idx = tid; idx < SROWS * SWID * CIN; idx += 256) {
    int pair = idx / SWID;           // j*32 + ci
    int wloc = idx - pair * SWID;    // 0..129
    int j  = pair >> 5;              // staged row 0..3
    int ci = pair & 31;
    int hr = h0 - 1 + j;
    int wg = w0 - 1 + wloc;
    float v = 0.f;
    if ((unsigned)hr < (unsigned)HSZ && (unsigned)wg < (unsigned)WSZ)
      v = x[(((size_t)n * CIN + ci) * HSZ + hr) * WSZ + wg];
    sX[(j * SWID + wloc) * XPITCH + ci] = f2bf(v);
  }

  __syncthreads();

  const int lane   = tid & 31;
  const int wid    = tid >> 5;
  const int cotile = wid & 1;     // which 16 output channels
  const int pq     = wid >> 1;    // which 32-pixel group (0..3)
  const int half   = lane >> 4;   // K-half selector
  const int col    = lane & 15;   // row (A) / column (B,C)

  v8f acc[HT][2] = {};            // [row][16-pixel tile] f32 accumulators

  const int corow = cotile * 16 + col;

#pragma unroll
  for (int kk = 0; kk < 9; ++kk) {
    const int kh = kk / 3;
    const int kw = kk - kh * 3;

    // A fragment: 16(co) x 32(ci) bf16; lane<16: K {0..7,16..23}, lane>=16: K {8..15,24..31}
    const unsigned short* pa = &sW[(kk * COUT + corow) * CIN + half * 8];
    v16bf a = frag16(pa, 16);

#pragma unroll
    for (int hloc = 0; hloc < HT; ++hloc) {
      const int j = hloc + kh;  // staged row index for this tap
#pragma unroll
      for (int t = 0; t < 2; ++t) {
        // B fragment: 32(ci) x 16(px); the (kw-1) shift implements the conv
        const int wloc = pq * 32 + t * 16 + col + kw;  // w - (w0-1) source index
        const unsigned short* pb = &sX[(j * SWID + wloc) * XPITCH + half * 8];
        v16bf b = frag16(pb, 16);
        acc[hloc][t] = __builtin_amdgcn_wmma_f32_16x16x32_bf16(
            false, a, false, b, (short)0, acc[hloc][t], false, false);
      }
    }
  }

  // ---- epilogue: scale by 1/sqrt(288) in f32 and store ----
  const float scale = 0.05892556509887896f;
  const int mbase = cotile * 16 + half * 8;
#pragma unroll
  for (int hloc = 0; hloc < HT; ++hloc) {
    const int hh = h0 + hloc;
#pragma unroll
    for (int t = 0; t < 2; ++t) {
      const int wcol = w0 + pq * 32 + t * 16 + col;
#pragma unroll
      for (int r = 0; r < 8; ++r) {
        const int co = mbase + r;
        out[(((size_t)n * COUT + co) * HSZ + hh) * WSZ + wcol] = acc[hloc][t][r] * scale;
      }
    }
  }
}

extern "C" void kernel_launch(void* const* d_in, const int* in_sizes, int n_in,
                              void* d_out, int out_size, void* d_ws, size_t ws_size,
                              hipStream_t stream) {
  const float* x   = (const float*)d_in[0];
  const float* wgt = (const float*)d_in[1];
  float* out       = (float*)d_out;
  dim3 grid(WSZ / WT, HSZ / HT, NBATCH);  // (2, 128, 32)
  bconv3x3_bin_wmma<<<grid, 256, 0, stream>>>(x, wgt, out);
}
